// SWMulitHeadAttention_10239202033876
// MI455X (gfx1250) — matrix operand order repaired and checked by reading, not compiled, not run on hardware
//
#include <hip/hip_runtime.h>
#include <hip/hip_bf16.h>
#include <math.h>

// ---------------------------------------------------------------------------
// Shifted-window MHA for MI455X (gfx1250, wave32, WMMA).
//   [qkv_kernel] fused roll(-3,-3) + 3x GEMM (f32 in -> bf16 out),
//                weight stripes staged transposed in LDS (b128 frag reads)
//   [attn_kernel] per-window 49x49 attention on WMMA; pos-bias staged via
//                global_load_async_to_lds (ASYNCcnt); V staged transposed in LDS
//   [fc_kernel]   GEMM + bias + fused roll(+3,+3), f32 out
// HBM-bound (~37 GFLOP vs ~400MB @ 23.3TB/s): bf16 intermediates, L2-friendly
// grid order, LDS-staged operands, wide b64/b128 fragment loads.
// ---------------------------------------------------------------------------

typedef __bf16 bf16;
typedef bf16  v16bf __attribute__((ext_vector_type(16)));
typedef float v8f   __attribute__((ext_vector_type(8)));

#define ROWS      100352            // 32*56*56
#define ROWTILES  6272              // ROWS/16
#define DMODEL    192
#define WIN_ELEMS 1568              // 49*32

union FragU {                       // bf16 fragment with wide-load views
  v16bf  bf;
  uint32_t u[8];
  uint4  q[2];
};

// A-fragment K interleave for 16-bit WMMA (ISA 7.12.2): VGPR j holds
// K = {0,2,4,6,16,18,20,22}[j] + 8*half + {0,1}
__device__ __forceinline__ int kmapA(int j) { return (j < 4) ? 2 * j : 8 + 2 * j; }

// ---------------------------------------------------------------------------
// Kernel 1: QKV. grid=(12, 784), block=256 (8 waves = 8 row-tiles).
// Weight column-stripes (16 cols x 192 rows) staged TRANSPOSED as bf16 in LDS
// once per block; B-fragments become 2x ds_load_b128.
// ---------------------------------------------------------------------------
__global__ __launch_bounds__(256) void qkv_kernel(
    const float* __restrict__ x,
    const float* __restrict__ Wq, const float* __restrict__ Wk,
    const float* __restrict__ Wv,
    bf16* __restrict__ qo, bf16* __restrict__ ko, bf16* __restrict__ vo)
{
  __shared__ __align__(16) bf16 s_wq[16][DMODEL];   // [n][k], 6KB each
  __shared__ __align__(16) bf16 s_wk[16][DMODEL];
  __shared__ __align__(16) bf16 s_wv[16][DMODEL];

  const int tid = threadIdx.x;
  const int colTile = blockIdx.x;       // 0..11 (fastest -> L2 reuse of x strip)

  {  // stage: thread t -> n = t%16, k rows t/16 + 16*it
    const int n   = tid & 15;
    const int kr0 = tid >> 4;
    const int gn  = colTile * 16 + n;
    for (int it = 0; it < 12; ++it) {
      const int kr = kr0 + 16 * it;
      s_wq[n][kr] = (bf16)Wq[kr * DMODEL + gn];
      s_wk[n][kr] = (bf16)Wk[kr * DMODEL + gn];
      s_wv[n][kr] = (bf16)Wv[kr * DMODEL + gn];
    }
  }
  __syncthreads();

  const int wave = tid >> 5, lane = tid & 31;
  const int m = lane & 15, half = lane >> 4;
  const int rowTile = blockIdx.y * 8 + wave;

  // fused cyclic shift roll(-3,-3): shifted[b,h,w] = x[b,(h+3)%56,(w+3)%56]
  const int gm = rowTile * 16 + m;
  const int b  = gm / 3136;
  const int s  = gm - b * 3136;
  const int hh = s / 56, ww = s - hh * 56;
  const int srcRow = b * 3136 + ((hh + 3) % 56) * 56 + ((ww + 3) % 56);
  const float2* xr2 = (const float2*)(x + (size_t)srcRow * DMODEL); // 768B-aligned
  const int n = colTile * 16 + m;

  v8f cq = {}, ck = {}, cv = {};
  for (int kk = 0; kk < 6; ++kk) {
    const int k0 = kk * 32;
    v16bf a;
#pragma unroll
    for (int j = 0; j < 8; ++j) {
      const float2 f = xr2[(k0 + 2 * j + 16 * half) >> 1];   // b64 load
      a[2 * j]     = (bf16)f.x;
      a[2 * j + 1] = (bf16)f.y;
    }
    // B-fragments: lane n=m holds k = k0 + e + 16*half contiguous -> b128 x2
    FragU bq, bk, bv;
    {
      const uint4* pq = (const uint4*)&s_wq[m][k0 + 16 * half];
      const uint4* pk = (const uint4*)&s_wk[m][k0 + 16 * half];
      const uint4* pv = (const uint4*)&s_wv[m][k0 + 16 * half];
      bq.q[0] = pq[0]; bq.q[1] = pq[1];
      bk.q[0] = pk[0]; bk.q[1] = pk[1];
      bv.q[0] = pv[0]; bv.q[1] = pv[1];
    }
    cq = __builtin_amdgcn_wmma_f32_16x16x32_bf16(false, a, false, bq.bf, (short)0, cq, false, false);
    ck = __builtin_amdgcn_wmma_f32_16x16x32_bf16(false, a, false, bk.bf, (short)0, ck, false, false);
    cv = __builtin_amdgcn_wmma_f32_16x16x32_bf16(false, a, false, bv.bf, (short)0, cv, false, false);
  }
#pragma unroll
  for (int r = 0; r < 8; ++r) {
    const int row = rowTile * 16 + r + 8 * half;
    const size_t idx = (size_t)row * DMODEL + n;
    qo[idx] = (bf16)cq[r];
    ko[idx] = (bf16)ck[r];
    vo[idx] = (bf16)cv[r];
  }
}

// ---------------------------------------------------------------------------
// Kernel 2: per-window attention. grid=12288 (=32*6*64), block=128 (4 waves).
// Wave w owns query-row strip [16w, 16w+16); 49 pads to 64.
// ---------------------------------------------------------------------------
__global__ __launch_bounds__(128) void attn_kernel(
    const bf16* __restrict__ q, const bf16* __restrict__ k,
    const bf16* __restrict__ v, const float* __restrict__ pe,
    bf16* __restrict__ ao)
{
  __shared__ float s_pe[169];
  __shared__ __align__(16) bf16 s_vt[32 * 64];  // V transposed [d][key], 4KB
  __shared__ __align__(16) bf16 s_p[64 * 64];   // softmax probs, 8KB

  const int win  = blockIdx.x;          // ((b*6+h)*64 + n)
  const int nwin = win & 63;            // window index within 64
  const int tid  = threadIdx.x;
  const int wave = tid >> 5;
  const int lane = tid & 31;
  const int m = lane & 15, half = lane >> 4;
  const size_t base = (size_t)win * WIN_ELEMS;

  // ---- stage pos_embedding via async global->LDS copy (ASYNCcnt path) ----
  for (int i = tid; i < 169; i += 128) {
    const uint32_t la = (uint32_t)(uintptr_t)(&s_pe[i]);  // flat LDS addr[31:0]
    const unsigned long long ga = (unsigned long long)(uintptr_t)(pe + i);
    asm volatile("global_load_async_to_lds_b32 %0, %1, off"
                 :: "v"(la), "v"(ga) : "memory");
  }
  // ---- stage V transposed: s_vt[d][key], keys 49..63 zero-padded ----
  for (int j = tid; j < 32 * 64; j += 128) {
    const int key = j >> 5, d = j & 31;               // d fastest -> coalesced
    s_vt[d * 64 + key] = (key < 49) ? v[base + (size_t)key * 32 + d] : (bf16)0.0f;
  }
  asm volatile("s_wait_asynccnt 0" ::: "memory");
  __syncthreads();

  const int qt = wave;

  // ---- scores = Q K^T : 4 WMMAs (K = d = 32 exactly) ----
  FragU aq;
  {
    const int t  = qt * 16 + m;
    const bool ok = (t < 49);
    const uint32_t* qr32 = (const uint32_t*)(q + base + (size_t)(ok ? t : 0) * 32);
#pragma unroll
    for (int j = 0; j < 8; ++j)
      aq.u[j] = ok ? qr32[(2 * j + 16 * half) >> 1] : 0u;   // dword loads
  }
  v8f acc[4] = {v8f{}, v8f{}, v8f{}, v8f{}};
#pragma unroll
  for (int nt = 0; nt < 4; ++nt) {
    const int key = nt * 16 + m;
    const bool ok = (key < 49);
    // per-lane 32B contiguous, 32B-aligned -> two b128 loads
    const uint4* kr4 = (const uint4*)(k + base + (size_t)(ok ? key : 0) * 32 + 16 * half);
    const uint4 z4 = make_uint4(0u, 0u, 0u, 0u);
    FragU bk;
    bk.q[0] = ok ? kr4[0] : z4;
    bk.q[1] = ok ? kr4[1] : z4;
    acc[nt] = __builtin_amdgcn_wmma_f32_16x16x32_bf16(false, aq.bf, false, bk.bf, (short)0, acc[nt], false, false);
  }

  // ---- scale + rel-pos bias + shift masks + row softmax ----
  const float scale = 5.656854249492381f;  // sqrt(32), reference MULTIPLIES
  float p[4][8];
#pragma unroll
  for (int r = 0; r < 8; ++r) {
    const int qrow = qt * 16 + r + 8 * half;
    const int qa = qrow / 7, qj = qrow % 7;
    float mx = -3.0e38f;
#pragma unroll
    for (int nt = 0; nt < 4; ++nt) {
      const int kc = nt * 16 + m;
      float val;
      if (kc < 49) {
        const int ka = kc / 7, kj = kc % 7;
        int ri = ka - qa + 6; ri = ri < 0 ? 0 : (ri > 12 ? 12 : ri);
        int rj = kj - qj + 6; rj = rj < 0 ? 0 : (rj > 12 ? 12 : rj);
        val = acc[nt][r] * scale + s_pe[ri * 13 + rj];
        if (nwin >= 56 && ((qrow >= 28) != (kc >= 28))) val = -1.0e30f;  // LOWER_MASK
        if (nwin == 7  && ((qj >= 4)   != (kj >= 4)))   val = -1.0e30f;  // RIGHT_MASK
      } else {
        val = -1.0e30f;   // pad columns
      }
      p[nt][r] = val;
      mx = fmaxf(mx, val);
    }
    // row lives in one 16-lane half-group -> xor 1,2,4,8 stays in-group
    for (int off = 1; off < 16; off <<= 1) mx = fmaxf(mx, __shfl_xor(mx, off, 32));
    float sum = 0.0f;
#pragma unroll
    for (int nt = 0; nt < 4; ++nt) {
      const float e = __expf(p[nt][r] - mx);
      p[nt][r] = e;
      sum += e;
    }
    for (int off = 1; off < 16; off <<= 1) sum += __shfl_xor(sum, off, 32);
    const float inv = 1.0f / sum;
#pragma unroll
    for (int nt = 0; nt < 4; ++nt) p[nt][r] *= inv;
  }

  // ---- re-shape probs C-layout -> A-layout via LDS (bf16) ----
#pragma unroll
  for (int nt = 0; nt < 4; ++nt)
#pragma unroll
    for (int r = 0; r < 8; ++r) {
      const int t = qt * 16 + r + 8 * half;
      const int c = nt * 16 + m;
      s_p[t * 64 + c] = (bf16)p[nt][r];
    }
  __syncthreads();

  // ---- out = P V : 2 K-steps x 2 N-tiles of WMMA ----
  v8f oacc[2] = {v8f{}, v8f{}};
#pragma unroll
  for (int ks = 0; ks < 2; ++ks) {
    FragU ap;
    {
      const int t = qt * 16 + m;
      const uint32_t* sp32 = (const uint32_t*)(s_p + t * 64);  // 128B-aligned rows
#pragma unroll
      for (int j = 0; j < 8; ++j)
        ap.u[j] = sp32[(ks * 32 + 2 * j + 16 * half) >> 1];  // ds_load_b32
    }
#pragma unroll
    for (int ntv = 0; ntv < 2; ++ntv) {
      const int d = ntv * 16 + m;
      // V fragment from transposed LDS: contiguous 16 bf16 -> 2x ds_load_b128
      FragU bv;
      const uint4* vp = (const uint4*)&s_vt[d * 64 + ks * 32 + 16 * half];
      bv.q[0] = vp[0];
      bv.q[1] = vp[1];
      oacc[ntv] = __builtin_amdgcn_wmma_f32_16x16x32_bf16(false, ap.bf, false, bv.bf, (short)0, oacc[ntv], false, false);
    }
  }
#pragma unroll
  for (int ntv = 0; ntv < 2; ++ntv) {
    const int d = ntv * 16 + m;
#pragma unroll
    for (int r = 0; r < 8; ++r) {
      const int t = qt * 16 + r + 8 * half;
      if (t < 49) ao[base + (size_t)t * 32 + d] = (bf16)oacc[ntv][r];
    }
  }
}

// ---------------------------------------------------------------------------
// Kernel 3: FC + bias + fused roll(+3,+3). grid=(12, 784), block=256.
// ---------------------------------------------------------------------------
__global__ __launch_bounds__(256) void fc_kernel(
    const bf16* __restrict__ a, const float* __restrict__ Wfc,
    const float* __restrict__ bfc, float* __restrict__ out)
{
  __shared__ __align__(16) bf16 s_w[16][DMODEL];   // [n][k] transposed stripe

  const int tid = threadIdx.x;
  const int colTile = blockIdx.x;

  {
    const int n   = tid & 15;
    const int kr0 = tid >> 4;
    const int gn  = colTile * 16 + n;
    for (int it = 0; it < 12; ++it) {
      const int kr = kr0 + 16 * it;
      s_w[n][kr] = (bf16)Wfc[kr * DMODEL + gn];
    }
  }
  __syncthreads();

  const int wave = tid >> 5, lane = tid & 31;
  const int m = lane & 15, half = lane >> 4;
  const int rowTile = blockIdx.y * 8 + wave;

  const uint32_t* ar32 = (const uint32_t*)(a + (size_t)(rowTile * 16 + m) * DMODEL);
  const int n = colTile * 16 + m;

  v8f c = {};
  for (int kk = 0; kk < 6; ++kk) {
    const int k0 = kk * 32;
    FragU af, bw;
#pragma unroll
    for (int j = 0; j < 8; ++j)
      af.u[j] = ar32[(k0 + 2 * j + 16 * half) >> 1];         // dword loads
    {
      const uint4* pw = (const uint4*)&s_w[m][k0 + 16 * half];
      bw.q[0] = pw[0];
      bw.q[1] = pw[1];
    }
    c = __builtin_amdgcn_wmma_f32_16x16x32_bf16(false, af.bf, false, bw.bf, (short)0, c, false, false);
  }
  const float bias = bfc[n];
#pragma unroll
  for (int r = 0; r < 8; ++r) {
    const int gm = rowTile * 16 + r + 8 * half;
    const int b  = gm / 3136;
    const int s  = gm - b * 3136;
    const int hh = s / 56, ww = s - hh * 56;
    // inverse roll: value computed at (h',w') lands at ((h'+3)%56,(w'+3)%56)
    const int dst = b * 3136 + ((hh + 3) % 56) * 56 + ((ww + 3) % 56);
    out[(size_t)dst * DMODEL + n] = c[r] + bias;
  }
}

// ---------------------------------------------------------------------------
extern "C" void kernel_launch(void* const* d_in, const int* in_sizes, int n_in,
                              void* d_out, int out_size, void* d_ws, size_t ws_size,
                              hipStream_t stream) {
  const float* x   = (const float*)d_in[0];
  const float* Wq  = (const float*)d_in[1];
  const float* Wk  = (const float*)d_in[2];
  const float* Wv  = (const float*)d_in[3];
  const float* Wfc = (const float*)d_in[4];
  const float* bfc = (const float*)d_in[5];
  const float* pe  = (const float*)d_in[6];

  const size_t E = (size_t)ROWS * DMODEL;   // 19,267,584 elements
  bf16* qo = (bf16*)d_ws;                   // bf16 intermediates: 4*E*2B ~ 154MB
  bf16* ko = qo + E;
  bf16* vo = ko + E;
  bf16* ao = vo + E;

  dim3 gGemm(12, ROWTILES / 8);             // colTile fastest for L2 reuse
  qkv_kernel<<<gGemm, 256, 0, stream>>>(x, Wq, Wk, Wv, qo, ko, vo);
  attn_kernel<<<12288, 128, 0, stream>>>(qo, ko, vo, pe, ao);
  fc_kernel<<<gGemm, 256, 0, stream>>>(ao, Wfc, bfc, (float*)d_out);
}
